// SingleObjectImplicitNetworkGrid_21766894256743
// MI455X (gfx1250) — matrix operand order repaired and checked by reading, not compiled, not run on hardware
//
#include <hip/hip_runtime.h>

// ---------- types ----------
typedef __attribute__((ext_vector_type(16))) _Float16 v16h;
typedef __attribute__((ext_vector_type(8)))  float    v8f;
typedef __attribute__((ext_vector_type(4)))  unsigned int u32x4v;
typedef __attribute__((ext_vector_type(8)))  int          i32x8v;
typedef __attribute__((ext_vector_type(4)))  int          i32x4v;

// ---------- problem constants ----------
#define NPTS       524288
#define NUM_LEVELS 16
#define TSIZE      (1u << 19)
#define PRIME1     2654435761u
#define PRIME2     805459861u

// grid resolutions: ceil(16 * 2^(7l/15))
__constant__ int c_res[NUM_LEVELS] = {16, 23, 31, 43, 59, 81, 112, 154, 213, 295,
                                      407, 562, 777, 1073, 1482, 2048};

// MLP dims (K padded to x32, N padded to x16)
#define K0 71
#define K0P 96
#define N0 256
#define N1 256
#define N2 257
#define N2P 272

// workspace layout (bytes)
#define W0F_OFF 0
#define W0F_BYTES (3 * 16 * 512 * 2)           // kT=3,nT=16, 512 halves/frag
#define W1F_OFF (W0F_OFF + W0F_BYTES)
#define W1F_BYTES (8 * 16 * 512 * 2)
#define W2F_OFF (W1F_OFF + W1F_BYTES)
#define W2F_BYTES (8 * 17 * 512 * 2)
#define X0_OFF (W2F_OFF + W2F_BYTES)           // N x 96 f16 encoded inputs

// ---------- helpers ----------
// Position of element (k,n) inside the pre-swizzled B-fragment buffer.
// B tile is [K=32][N=16]; lane = n%16 + 16*(krem>=16); within lane 16 halves:
// vgpr v holds K = base+2v, base+2v+1 (base = 16*(lane>=16)).
__device__ __forceinline__ int frag_half_index(int nT, int kt, int nt, int krem, int nrem) {
  int hi   = (krem >= 16) ? 1 : 0;
  int r2   = krem - hi * 16;
  int lane = nrem + hi * 16;
  return (((kt * nT + nt) * 32) + lane) * 16 + (r2 >> 1) * 2 + (r2 & 1);
}

// A fragment (16x32 f16), matching B-fragment K order: lane<16 -> row M=lane, K {0..15};
// lane>=16 -> row M=lane-16, K {16..31}. Two 16B chunks, contiguous in K.
__device__ __forceinline__ v16h load_a_frag(const _Float16* row, int kt, int hi) {
  union { v16h v; uint4 q[2]; } f;
  int k0 = kt * 32 + hi * 16;
  f.q[0] = *(const uint4*)(row + k0);
  f.q[1] = *(const uint4*)(row + k0 + 8);
  return f.v;
}

__device__ __forceinline__ v16h load_b_frag(const _Float16* wf, int kt, int nt, int nT, int lane) {
  const _Float16* p = wf + (((size_t)kt * nT + nt) * 32 + lane) * 16;
  __builtin_prefetch(p + 512, 0, 1);   // global_prefetch of next fragment
  union { v16h v; uint4 q[2]; } f;
  f.q[0] = *(const uint4*)p;
  f.q[1] = *(const uint4*)(p + 8);
  return f.v;
}

__device__ __forceinline__ v8f wmma_f16(v16h a, v16h b, v8f c) {
  return __builtin_amdgcn_wmma_f32_16x16x32_f16(false, a, false, b, (short)0, c, false, false);
}

// softplus(100x)/100, numerically stable
__device__ __forceinline__ float softplus100(float x) {
  return fmaxf(x, 0.f) + __logf(1.f + __expf(-fabsf(100.f * x))) * 0.01f;
}

// ---------- kernel 1: weight-norm + f16 swizzle into WMMA B-fragment layout ----------
__global__ __launch_bounds__(256) void prep_weights(
    const float* __restrict__ v0, const float* __restrict__ g0,
    const float* __restrict__ v1, const float* __restrict__ g1,
    const float* __restrict__ v2, const float* __restrict__ g2,
    _Float16* __restrict__ w0f, _Float16* __restrict__ w1f, _Float16* __restrict__ w2f) {
  __shared__ float red[256];
  int b = blockIdx.x;
  const float *V, *G; _Float16* W;
  int K, Kp, n, Np, realN;
  if (b < 256)      { V = v0; G = g0; W = w0f; K = K0;  Kp = K0P; n = b;       Np = 256; realN = 256; }
  else if (b < 512) { V = v1; G = g1; W = w1f; K = 256; Kp = 256; n = b - 256; Np = 256; realN = 256; }
  else              { V = v2; G = g2; W = w2f; K = 256; Kp = 256; n = b - 512; Np = N2P; realN = N2; }
  int nT = Np / 16;

  float scale = 0.f;
  if (n < realN) {
    float s = 0.f;
    for (int k = threadIdx.x; k < K; k += 256) { float t = V[n * K + k]; s += t * t; }
    red[threadIdx.x] = s;
    __syncthreads();
    for (int off = 128; off > 0; off >>= 1) {
      if ((int)threadIdx.x < off) red[threadIdx.x] += red[threadIdx.x + off];
      __syncthreads();
    }
    scale = G[n] * rsqrtf(red[0] + 1e-30f);
  }
  int nt = n >> 4, nr = n & 15;
  for (int k = threadIdx.x; k < Kp; k += 256) {
    float val = (n < realN && k < K) ? V[n * K + k] * scale : 0.f;
    W[frag_half_index(nT, k >> 5, nt, k & 31, nr)] = (_Float16)val;
  }
}

// ---------- kernel 2: pos-embed + multires hash encode -> f16 rows [N][96] ----------
__global__ __launch_bounds__(256) void encode_kernel(
    const float* __restrict__ x, const float* __restrict__ tables,
    _Float16* __restrict__ X0) {
  int i = blockIdx.x * 256 + threadIdx.x;
  if (i >= NPTS) return;
  float p[3];
#pragma unroll
  for (int d = 0; d < 3; d++) p[d] = fminf(fmaxf(x[i * 3 + d], 0.f), 1.f);

  _Float16* r = X0 + (size_t)i * K0P;
#pragma unroll
  for (int d = 0; d < 3; d++) r[d] = (_Float16)p[d];
  int o = 3;
#pragma unroll
  for (int j = 0; j < 6; j++) {
    float f = (float)(1 << j);
#pragma unroll
    for (int d = 0; d < 3; d++) r[o + d]     = (_Float16)__sinf(f * p[d]);
#pragma unroll
    for (int d = 0; d < 3; d++) r[o + 3 + d] = (_Float16)__cosf(f * p[d]);
    o += 6;
  }
  // o == 39
  for (int l = 0; l < NUM_LEVELS; l++) {
    int R = c_res[l];
    float w[3]; unsigned u0[3];
#pragma unroll
    for (int d = 0; d < 3; d++) {
      float pf = p[d] * (float)R;
      float fl = floorf(pf);
      w[d]  = pf - fl;
      u0[d] = (unsigned)fl;
    }
    float a0 = 0.f, a1 = 0.f;
    bool dense = (l < 5);  // (R+1)^3 <= 2^19 holds exactly for levels 0..4
#pragma unroll
    for (int c = 0; c < 8; c++) {
      unsigned id[3];
#pragma unroll
      for (int d = 0; d < 3; d++) {
        unsigned t = u0[d] + ((unsigned)(c >> d) & 1u);
        id[d] = t > (unsigned)R ? (unsigned)R : t;
      }
      unsigned flat;
      if (dense) flat = id[0] + (unsigned)(R + 1) * (id[1] + (unsigned)(R + 1) * id[2]);
      else       flat = (id[0] ^ (id[1] * PRIME1) ^ (id[2] * PRIME2)) & (TSIZE - 1u);
      const float* tp = tables + ((size_t)l * TSIZE + flat) * 2;
      float wc = ((c & 1) ? w[0] : 1.f - w[0]) *
                 ((c & 2) ? w[1] : 1.f - w[1]) *
                 ((c & 4) ? w[2] : 1.f - w[2]);
      a0 += wc * tp[0];
      a1 += wc * tp[1];
    }
    r[39 + 2 * l]     = (_Float16)a0;
    r[39 + 2 * l + 1] = (_Float16)a1;
  }
#pragma unroll
  for (int k = K0; k < K0P; k++) r[k] = (_Float16)0.f;
}

// ---------- kernel 3: fused 3-layer WMMA MLP ----------
// Block: 256 threads (8 waves), 128 points. TDM loads the 128x96 f16 input tile
// into LDS; each wave owns 16 rows through all three layers (inter-layer
// activations live in per-wave LDS regions; no HBM round-trips).
__global__ __launch_bounds__(256) void mlp_kernel(
    const _Float16* __restrict__ X0,
    const _Float16* __restrict__ w0f, const _Float16* __restrict__ w1f,
    const _Float16* __restrict__ w2f,
    const float* __restrict__ b0, const float* __restrict__ b1,
    const float* __restrict__ b2,
    float* __restrict__ out) {
  extern __shared__ char smem[];
  _Float16* sX = (_Float16*)smem;                 // 128*96 halves = 24576 B, at LDS addr 0
  _Float16* sH = (_Float16*)(smem + 128 * K0P * 2);

  const int wave = threadIdx.x >> 5;
  const int lane = threadIdx.x & 31;
  const int ln   = lane & 15;   // output column within a 16-wide tile
  const int hi   = lane >> 4;   // row-half selector for C/D layout

  _Float16* H1 = sH + wave * (16 * 256 * 2);
  _Float16* H2 = H1 + 16 * 256;

  // --- async TDM: stage [128 x 96] f16 tile (row-major, contiguous) into LDS ---
  if (wave == 0) {
    unsigned long long ga =
        (unsigned long long)(const void*)(X0 + (size_t)blockIdx.x * 128 * K0P);
    u32x4v g0;
    g0.x = 1u;                                   // count=1, is_restore=0, gather=0
    g0.y = 0u;                                   // lds_addr = 0 (tile at LDS base)
    g0.z = (unsigned)(ga & 0xFFFFFFFFu);         // global_addr[31:0]
    g0.w = (unsigned)((ga >> 32) & 0x01FFFFFFu) | (2u << 30);  // addr[56:32] | type=2
    i32x8v g1;
    g1.s0 = 0x00010000;          // workgroup_mask=0, data_size=1 (2B)
    g1.s1 = (96 << 16);          // tensor_dim0 = 96 (bits 79:48)
    g1.s2 = (128 << 16);         // tensor_dim1 = 128 (bits 111:80)
    g1.s3 = (96 << 16);          // tile_dim0 = 96 (bits 127:112)
    g1.s4 = 128;                 // tile_dim1 = 128 (bits 143:128)
    g1.s5 = 96;                  // tensor_dim0_stride = 96
    g1.s6 = 0;
    g1.s7 = 0;
    i32x4v z4 = {0, 0, 0, 0};               // groups 2/3 unused (2D tensor)
    i32x8v z8 = {0, 0, 0, 0, 0, 0, 0, 0};   // extra group (6-arg builtin form)
    __builtin_amdgcn_tensor_load_to_lds(g0, g1, z4, z4, z8, 0);
    __builtin_amdgcn_s_wait_tensorcnt(0);
  }
  __syncthreads();

  // ---- layer 0: [16x96] @ [96x256] + b0, softplus100 -> H1 ----
  {
    const _Float16* aRow = sX + (size_t)(wave * 16 + ln) * K0P;
    for (int nt = 0; nt < 16; nt++) {
      float bias = b0[nt * 16 + ln];
      v8f acc = {bias, bias, bias, bias, bias, bias, bias, bias};
#pragma unroll
      for (int kt = 0; kt < 3; kt++)
        acc = wmma_f16(load_a_frag(aRow, kt, hi), load_b_frag(w0f, kt, nt, 16, lane), acc);
#pragma unroll
      for (int rr = 0; rr < 8; rr++)
        H1[(rr + hi * 8) * 256 + nt * 16 + ln] = (_Float16)softplus100(acc[rr]);
    }
  }

  // ---- layer 1: [16x256] @ [256x256] + b1, softplus100 -> H2 ----
  {
    const _Float16* aRow = H1 + (size_t)ln * 256;
    for (int nt = 0; nt < 16; nt++) {
      float bias = b1[nt * 16 + ln];
      v8f acc = {bias, bias, bias, bias, bias, bias, bias, bias};
#pragma unroll
      for (int kt = 0; kt < 8; kt++)
        acc = wmma_f16(load_a_frag(aRow, kt, hi), load_b_frag(w1f, kt, nt, 16, lane), acc);
#pragma unroll
      for (int rr = 0; rr < 8; rr++)
        H2[(rr + hi * 8) * 256 + nt * 16 + ln] = (_Float16)softplus100(acc[rr]);
    }
  }

  // ---- layer 2: [16x256] @ [256x272] + b2 -> out[N][257] (f32) ----
  {
    const _Float16* aRow = H2 + (size_t)ln * 256;
    size_t rowBase = (size_t)blockIdx.x * 128 + (size_t)wave * 16;
    for (int nt = 0; nt < 17; nt++) {
      int n = nt * 16 + ln;
      float bias = (n < N2) ? b2[n] : 0.f;
      v8f acc = {bias, bias, bias, bias, bias, bias, bias, bias};
#pragma unroll
      for (int kt = 0; kt < 8; kt++)
        acc = wmma_f16(load_a_frag(aRow, kt, hi), load_b_frag(w2f, kt, nt, 17, lane), acc);
      if (n < N2) {
#pragma unroll
        for (int rr = 0; rr < 8; rr++)
          out[(rowBase + rr + hi * 8) * N2 + n] = acc[rr];
      }
    }
  }
}

// ---------- host launch ----------
extern "C" void kernel_launch(void* const* d_in, const int* in_sizes, int n_in,
                              void* d_out, int out_size, void* d_ws, size_t ws_size,
                              hipStream_t stream) {
  const float* x      = (const float*)d_in[0];
  const float* tables = (const float*)d_in[1];
  const float* v0 = (const float*)d_in[2];
  const float* g0 = (const float*)d_in[3];
  const float* b0 = (const float*)d_in[4];
  const float* v1 = (const float*)d_in[5];
  const float* g1 = (const float*)d_in[6];
  const float* b1 = (const float*)d_in[7];
  const float* v2 = (const float*)d_in[8];
  const float* g2 = (const float*)d_in[9];
  const float* b2 = (const float*)d_in[10];
  float* out = (float*)d_out;

  char* ws = (char*)d_ws;
  _Float16* w0f = (_Float16*)(ws + W0F_OFF);
  _Float16* w1f = (_Float16*)(ws + W1F_OFF);
  _Float16* w2f = (_Float16*)(ws + W2F_OFF);
  _Float16* X0  = (_Float16*)(ws + X0_OFF);

  prep_weights<<<256 + 256 + N2P, 256, 0, stream>>>(v0, g0, v1, g1, v2, g2, w0f, w1f, w2f);
  encode_kernel<<<NPTS / 256, 256, 0, stream>>>(x, tables, X0);

  // LDS: 24576 (input tile) + 8 waves * 2 * 16*256*2 (H1/H2) = 155648 B
  mlp_kernel<<<NPTS / 128, 256, 24576 + 8 * 2 * 16 * 256 * 2, stream>>>(
      X0, w0f, w1f, w2f, b0, b1, b2, out);
}
